// AttentionFusion_57097295233291
// MI455X (gfx1250) — matrix operand order repaired and checked by reading, not compiled, not run on hardware
//
#include <hip/hip_runtime.h>

#define BB 4
#define AA 3
#define SS 1024
#define HH 768
#define NHEAD 12
#define HD 64
#define LN_EPS 1e-5f

typedef _Float16 half_t;
typedef __attribute__((ext_vector_type(16))) _Float16 v16h;
typedef __attribute__((ext_vector_type(8)))  _Float16 v8h;
typedef __attribute__((ext_vector_type(8)))  float    v8f;

union V16 { v16h v; v8h h[2]; };

__device__ __forceinline__ v8f wmma_f16(v16h a, v16h b, v8f c) {
  // D = A(16x32 f16) * B(32x16 f16) + C(16x16 f32)
  return __builtin_amdgcn_wmma_f32_16x16x32_f16(false, a, false, b, (short)0, c,
                                                false, false);
}

// ---------------------------------------------------------------- utilities
__global__ __launch_bounds__(256) void cvt_f16_k(const float* __restrict__ in,
                                                 half_t* __restrict__ out, int n) {
  int i = blockIdx.x * 256 + threadIdx.x;
  if (i < n) out[i] = (half_t)in[i];
}

__global__ __launch_bounds__(256) void fill_k(float* __restrict__ p, size_t n, float v) {
  size_t i = (size_t)blockIdx.x * 256 + threadIdx.x;
  size_t stride = (size_t)gridDim.x * 256;
  for (; i < n; i += stride) p[i] = v;
}

// ---------------------------------------------------------------- GEMM: Y = X * W^T + b
// X: f32 [rows, 768]; W: f16 [768, 768] (row-major, W[n][k]); block tile 128x64,
// 8 waves, each wave owns a 16x64 strip (4 wmma accumulators), K-step 32.
// MODE 0: f16 out, natural [row][col]
// MODE 1: f16 out, V-transposed [b][h][a][d][s]
// MODE 2: f32 out, natural, + residual (output projection)
template <int MODE>
__global__ __launch_bounds__(256) void gemm_xwT_k(const float* __restrict__ X,
                                                  const half_t* __restrict__ W,
                                                  const float* __restrict__ bias,
                                                  const float* __restrict__ resid,
                                                  void* __restrict__ Yv) {
  __shared__ half_t Xs[128 * 40];
  __shared__ half_t Ws[64 * 40];
  const int tid  = threadIdx.x;
  const int lane = tid & 31;
  const int wave = tid >> 5;
  const int ln15 = lane & 15;
  const int lhi  = lane >> 4;      // 0/1
  const int p8   = lhi * 8;
  const int bn   = blockIdx.x;     // 0..11  (64 output cols each)
  const int bm   = blockIdx.y;     // 128 rows each

  v8f c[4];
  const v8f vzero = {0.f, 0.f, 0.f, 0.f, 0.f, 0.f, 0.f, 0.f};
#pragma unroll
  for (int i = 0; i < 4; ++i) c[i] = vzero;

  for (int k0 = 0; k0 < HH; k0 += 32) {
    // stage X tile (128x32 f32 -> f16)
#pragma unroll
    for (int it = 0; it < 4; ++it) {
      const int r  = it * 32 + (tid >> 3);
      const int cg = (tid & 7) * 4;
      const float4 f = *(const float4*)&X[(size_t)(bm * 128 + r) * HH + k0 + cg];
      Xs[r * 40 + cg + 0] = (half_t)f.x;
      Xs[r * 40 + cg + 1] = (half_t)f.y;
      Xs[r * 40 + cg + 2] = (half_t)f.z;
      Xs[r * 40 + cg + 3] = (half_t)f.w;
      if (k0 + 32 < HH)  // prefetch next K-step of X into cache hierarchy
        __builtin_prefetch(&X[(size_t)(bm * 128 + r) * HH + k0 + 32 + cg], 0, 1);
    }
    // stage W tile (64x32 f16)
    {
      const int r  = tid >> 2;
      const int cg = (tid & 3) * 8;
      *(v8h*)&Ws[r * 40 + cg] =
          *(const v8h*)&W[(size_t)(bn * 64 + r) * HH + k0 + cg];
    }
    __syncthreads();

    // A fragment: 16x32, lane<16: row=lane, k={0..15}; lane>=16: k={16..31}
    V16 af;
    const half_t* xr = &Xs[(wave * 16 + ln15) * 40];
    af.h[0] = *(const v8h*)(xr + lhi * 16);
    af.h[1] = *(const v8h*)(xr + lhi * 16 + 8);

#pragma unroll
    for (int nt = 0; nt < 4; ++nt) {
      // B fragment: B[k][n]=W[n][k]; lane group selects k-half, 32B contiguous
      V16 bf;
      const half_t* wr = &Ws[(nt * 16 + ln15) * 40 + lhi * 16];
      bf.h[0] = *(const v8h*)(wr);
      bf.h[1] = *(const v8h*)(wr + 8);
      c[nt]   = wmma_f16(af.v, bf.v, c[nt]);
    }
    __syncthreads();
  }

  // epilogue: D layout -> lane L VGPR r holds (m = r + 8*(L>=16), n = L%16)
#pragma unroll
  for (int nt = 0; nt < 4; ++nt) {
    const int gn  = bn * 64 + nt * 16 + ln15;
    const float bv = bias[gn];
#pragma unroll
    for (int r = 0; r < 8; ++r) {
      const int gm  = bm * 128 + wave * 16 + r + p8;
      const float val = c[nt][r] + bv;
      if (MODE == 0) {
        ((half_t*)Yv)[(size_t)gm * HH + gn] = (half_t)val;
      } else if (MODE == 1) {
        const int b = gm / (AA * SS);
        const int a = (gm / SS) % AA;
        const int s = gm % SS;
        const int h = gn >> 6;
        const int d = gn & 63;
        ((half_t*)Yv)[(size_t)(((b * NHEAD + h) * AA + a) * HD + d) * SS + s] =
            (half_t)val;
      } else {
        ((float*)Yv)[(size_t)gm * HH + gn] = val + resid[(size_t)gm * HH + gn];
      }
    }
  }
}

// ---------------------------------------------------------------- attention
// One wave per 16-row q tile of one (b, h). Per 32-wide s-chunk:
//   scores  S_a = Q K_a^T (12 wmma),  softmax over the adapter axis (raw
//   v_exp_f32 / v_rcp_f32 — argument <= 0, FTZ below 2^-126 is exactly what
//   softmax wants), P round-trips per-wave LDS (D-layout -> A-frag layout),
//   acc    += P_a V_a (12 wmma).
// Q16:  [b][q][h*64+d]  f16 ;  K16: [b][a][s][h*64+d] f16 ; Vt16: [b][h][a][d][s] f16
__global__ __launch_bounds__(256) void attn_k(const half_t* __restrict__ Q16,
                                              const half_t* __restrict__ K16,
                                              const half_t* __restrict__ Vt16,
                                              float* __restrict__ att) {
  __shared__ half_t Plds[8 * 16 * 40];
  const int tid  = threadIdx.x;
  const int lane = tid & 31;
  const int wave = tid >> 5;
  const int ln15 = lane & 15;
  const int lhi  = lane >> 4;
  const int p8   = lhi * 8;

  const int blk = blockIdx.x;
  const int b   = blk / (NHEAD * 8);
  const int h   = (blk / 8) % NHEAD;
  const int qb  = blk & 7;
  const int q0  = qb * 128 + wave * 16;

  // Q A-fragments for d-halves 0 (k=d 0..31) and 1 (k=d 32..63), held in VGPRs
  V16 qf[2];
  {
    const half_t* qbase = Q16 + (size_t)(b * SS + q0 + ln15) * HH + h * HD;
#pragma unroll
    for (int dh = 0; dh < 2; ++dh) {
      qf[dh].h[0] = *(const v8h*)(qbase + dh * 32 + lhi * 16);
      qf[dh].h[1] = *(const v8h*)(qbase + dh * 32 + lhi * 16 + 8);
    }
  }

  const v8f vzero = {0.f, 0.f, 0.f, 0.f, 0.f, 0.f, 0.f, 0.f};
  v8f acc[4];
#pragma unroll
  for (int i = 0; i < 4; ++i) acc[i] = vzero;

  half_t* pl = &Plds[wave * 16 * 40];
  const float kscale = 0.125f * 1.44269504f;  // (1/sqrt(HD)) * log2(e)

  for (int s0 = 0; s0 < SS; s0 += 32) {
    v8f sc[3][2];
#pragma unroll
    for (int a = 0; a < 3; ++a) {
      const half_t* kb = K16 + (size_t)((b * AA + a) * SS) * HH + h * HD;
#pragma unroll
      for (int st = 0; st < 2; ++st) {
        // B-frag of K^T: lane n = s (32B contiguous d per lane)
        const half_t* kr = kb + (size_t)(s0 + st * 16 + ln15) * HH + lhi * 16;
        V16 k0f, k1f;
        k0f.h[0] = *(const v8h*)(kr);
        k0f.h[1] = *(const v8h*)(kr + 8);
        k1f.h[0] = *(const v8h*)(kr + 32);
        k1f.h[1] = *(const v8h*)(kr + 40);
        v8f z = vzero;
        z = wmma_f16(qf[0].v, k0f.v, z);
        z = wmma_f16(qf[1].v, k1f.v, z);
        sc[a][st] = z;
      }
    }

    // softmax over the 3 adapters, elementwise; scale folded into exp2 arg.
    // __builtin_amdgcn_exp2f/rcpf map straight to v_exp_f32/v_rcp_f32 with no
    // libm range-fixup code (arg <= 0, FTZ underflow is the desired behavior).
#pragma unroll
    for (int st = 0; st < 2; ++st) {
#pragma unroll
      for (int e = 0; e < 8; ++e) {
        const float x0 = sc[0][st][e], x1 = sc[1][st][e], x2 = sc[2][st][e];
        const float m  = fmaxf(x0, fmaxf(x1, x2));
        const float e0 = __builtin_amdgcn_exp2f((x0 - m) * kscale);
        const float e1 = __builtin_amdgcn_exp2f((x1 - m) * kscale);
        const float e2 = __builtin_amdgcn_exp2f((x2 - m) * kscale);
        const float inv = __builtin_amdgcn_rcpf(e0 + e1 + e2);
        sc[0][st][e] = e0 * inv;
        sc[1][st][e] = e1 * inv;
        sc[2][st][e] = e2 * inv;
      }
    }

#pragma unroll
    for (int a = 0; a < 3; ++a) {
      // D-layout -> A-fragment layout via per-wave LDS scratch (in-order DS, no barrier)
#pragma unroll
      for (int st = 0; st < 2; ++st)
#pragma unroll
        for (int r = 0; r < 8; ++r)
          pl[(r + p8) * 40 + st * 16 + ln15] = (half_t)sc[a][st][r];

      V16 pa;
      const half_t* prow = pl + ln15 * 40;
      pa.h[0] = *(const v8h*)(prow + lhi * 16);
      pa.h[1] = *(const v8h*)(prow + lhi * 16 + 8);

      const half_t* vb = Vt16 + (size_t)(((b * NHEAD + h) * AA + a) * HD) * SS;
#pragma unroll
      for (int dt = 0; dt < 4; ++dt) {
        // B-frag of V: lane n = d, k = s (32B contiguous s per lane, thanks to Vt layout)
        const half_t* vr = vb + (size_t)(dt * 16 + ln15) * SS + s0 + lhi * 16;
        V16 vf;
        vf.h[0] = *(const v8h*)(vr);
        vf.h[1] = *(const v8h*)(vr + 8);
        acc[dt] = wmma_f16(pa.v, vf.v, acc[dt]);
      }
    }
  }

  // write attended tile (f32) back in natural [b][q][h*64+d] layout
  float* ob = att + (size_t)(b * SS + q0) * HH + h * HD;
#pragma unroll
  for (int dt = 0; dt < 4; ++dt)
#pragma unroll
    for (int r = 0; r < 8; ++r)
      ob[(size_t)(r + p8) * HH + dt * 16 + ln15] = acc[dt][r];
}

// ---------------------------------------------------------------- LayerNorm per row of 768
__global__ __launch_bounds__(256) void ln_k(const float* __restrict__ x,
                                            const float* __restrict__ gamma,
                                            const float* __restrict__ beta,
                                            float* __restrict__ out) {
  __shared__ float ssum[256];
  __shared__ float ssq[256];
  const int row = blockIdx.x;
  const int t   = threadIdx.x;
  const float* xr = x + (size_t)row * HH;
  const float v0 = xr[t], v1 = xr[t + 256], v2 = xr[t + 512];
  ssum[t] = v0 + v1 + v2;
  ssq[t]  = v0 * v0 + v1 * v1 + v2 * v2;
  __syncthreads();
#pragma unroll
  for (int o = 128; o > 0; o >>= 1) {
    if (t < o) { ssum[t] += ssum[t + o]; ssq[t] += ssq[t + o]; }
    __syncthreads();
  }
  const float mu  = ssum[0] * (1.0f / HH);
  const float var = ssq[0] * (1.0f / HH) - mu * mu;
  const float inv = rsqrtf(var + LN_EPS);
  float* orow = out + (size_t)row * HH;
  orow[t]       = (v0 - mu) * inv * gamma[t]       + beta[t];
  orow[t + 256] = (v1 - mu) * inv * gamma[t + 256] + beta[t + 256];
  orow[t + 512] = (v2 - mu) * inv * gamma[t + 512] + beta[t + 512];
}

// ---------------------------------------------------------------- host launcher
extern "C" void kernel_launch(void* const* d_in, const int* in_sizes, int n_in,
                              void* d_out, int out_size, void* d_ws, size_t ws_size,
                              hipStream_t stream) {
  (void)in_sizes; (void)n_in; (void)out_size; (void)ws_size;

  const float* query = (const float*)d_in[0];
  const float* adap  = (const float*)d_in[1];
  const float* Wq = (const float*)d_in[2];  const float* bq = (const float*)d_in[3];
  const float* Wk = (const float*)d_in[4];  const float* bk = (const float*)d_in[5];
  const float* Wv = (const float*)d_in[6];  const float* bv = (const float*)d_in[7];
  const float* Wo = (const float*)d_in[8];  const float* bo = (const float*)d_in[9];
  const float* gamma = (const float*)d_in[10];
  const float* beta  = (const float*)d_in[11];

  char* ws = (char*)d_ws;
  const size_t WBYTES = (size_t)HH * HH * sizeof(half_t);
  half_t* Wq16 = (half_t*)(ws + 0 * WBYTES);
  half_t* Wk16 = (half_t*)(ws + 1 * WBYTES);
  half_t* Wv16 = (half_t*)(ws + 2 * WBYTES);
  half_t* Wo16 = (half_t*)(ws + 3 * WBYTES);
  size_t off = 4 * WBYTES;
  half_t* Q16 = (half_t*)(ws + off);  off += (size_t)BB * SS * HH * sizeof(half_t);
  half_t* K16 = (half_t*)(ws + off);  off += (size_t)BB * AA * SS * HH * sizeof(half_t);
  half_t* Vt16 = (half_t*)(ws + off); off += (size_t)BB * AA * SS * HH * sizeof(half_t);
  float* att32 = (float*)(ws + off);  off += (size_t)BB * SS * HH * sizeof(float);
  float* x32   = (float*)(ws + off);

  // 1) weights f32 -> f16
  const int wn = HH * HH;
  cvt_f16_k<<<dim3((wn + 255) / 256), 256, 0, stream>>>(Wq, Wq16, wn);
  cvt_f16_k<<<dim3((wn + 255) / 256), 256, 0, stream>>>(Wk, Wk16, wn);
  cvt_f16_k<<<dim3((wn + 255) / 256), 256, 0, stream>>>(Wv, Wv16, wn);
  cvt_f16_k<<<dim3((wn + 255) / 256), 256, 0, stream>>>(Wo, Wo16, wn);

  // 2) projections (WMMA GEMMs)
  gemm_xwT_k<0><<<dim3(HH / 64, (BB * SS) / 128), 256, 0, stream>>>(
      query, Wq16, bq, nullptr, (void*)Q16);
  gemm_xwT_k<0><<<dim3(HH / 64, (BB * AA * SS) / 128), 256, 0, stream>>>(
      adap, Wk16, bk, nullptr, (void*)K16);
  gemm_xwT_k<1><<<dim3(HH / 64, (BB * AA * SS) / 128), 256, 0, stream>>>(
      adap, Wv16, bv, nullptr, (void*)Vt16);

  // 3) fused adapter attention (softmax over adapters), WMMA QK^T and PV
  attn_k<<<dim3(BB * NHEAD * 8), 256, 0, stream>>>(Q16, K16, Vt16, att32);

  // 4) output projection + bias + residual (WMMA GEMM, f32 out)
  gemm_xwT_k<2><<<dim3(HH / 64, (BB * SS) / 128), 256, 0, stream>>>(
      att32, Wo16, bo, query, (void*)x32);

  // 5) LayerNorm -> d_out[0 .. B*S*H)
  ln_k<<<dim3(BB * SS), 256, 0, stream>>>(x32, gamma, beta, (float*)d_out);

  // 6) avg_weights is analytically 1/A everywhere (softmax over adapter axis sums to 1)
  float* avg = (float*)d_out + (size_t)BB * SS * HH;
  fill_k<<<dim3(4096), 256, 0, stream>>>(avg, (size_t)BB * SS * SS, 1.0f / 3.0f);
}
